// Cat_20916490732013
// MI455X (gfx1250) — matrix operand order repaired and checked by reading, not compiled, not run on hardware
//
#include <hip/hip_runtime.h>

// ---------------------------------------------------------------------------
// Problem constants (from the reference)
// ---------------------------------------------------------------------------
#define NUM_STARTS   4096
#define NUM_CLASSES  32768
#define EMB_DIM      128
#define FDIM         256      // FEATURE_DIM

#define LN2   0.6931471805599453f
#define LOG2E 1.4426950408889634f

typedef __attribute__((ext_vector_type(16))) _Float16 v16h;
typedef __attribute__((ext_vector_type(8)))  float    v8f;

// ---------------------------------------------------------------------------
// Kernel 1: features.  For each row i of emb[N,128]:
//   f[t] = sum_k emb[i,k] * proj[k,t]   (t = 0..255)
//   m    = max_t f[t]
//   expX[i,t] = (f16) exp(f[t] - m),  mX[i] = m
// ---------------------------------------------------------------------------
__global__ __launch_bounds__(256)
void feat_kernel(const float* __restrict__ emb,
                 const float* __restrict__ proj,
                 _Float16* __restrict__ expX,
                 float* __restrict__ mX)
{
    const int tid = threadIdx.x;
    const int row = blockIdx.x;

    __shared__ float row_s[EMB_DIM];
    __shared__ float red[256];

    if (tid < EMB_DIM) row_s[tid] = emb[row * EMB_DIM + tid];
    __syncthreads();

    float acc = 0.0f;
    #pragma unroll 8
    for (int k = 0; k < EMB_DIM; ++k)
        acc = fmaf(row_s[k], proj[k * FDIM + tid], acc);

    red[tid] = acc;
    __syncthreads();
    #pragma unroll
    for (int s = 128; s >= 1; s >>= 1) {
        if (tid < s) red[tid] = fmaxf(red[tid], red[tid + s]);
        __syncthreads();
    }
    const float m = red[0];

    expX[row * FDIM + tid] = (_Float16)__expf(acc - m);
    if (tid == 0) mX[row] = m;
}

// ---------------------------------------------------------------------------
// Kernel 2: mMax = max_j mR[j]  (single block)
// ---------------------------------------------------------------------------
__global__ __launch_bounds__(256)
void mmax_kernel(const float* __restrict__ mR, float* __restrict__ mMax)
{
    __shared__ float red[256];
    const int tid = threadIdx.x;
    float v = -3.0e38f;
    for (int i = tid; i < NUM_CLASSES; i += 256) v = fmaxf(v, mR[i]);
    red[tid] = v;
    __syncthreads();
    #pragma unroll
    for (int s = 128; s >= 1; s >>= 1) {
        if (tid < s) red[tid] = fmaxf(red[tid], red[tid + s]);
        __syncthreads();
    }
    if (tid == 0) mMax[0] = red[0];
}

// ---------------------------------------------------------------------------
// Async global -> LDS copy of one 16x256 f16 tile (8 KB) via the CDNA5
// ASYNCcnt path (GLOBAL_LOAD_ASYNC_TO_LDS_B128): no VGPR round trip; the DMA
// overlaps with WMMA compute on the other buffer.
// ---------------------------------------------------------------------------
__device__ __forceinline__ void async_tile(const _Float16* __restrict__ gsrc,
                                           _Float16* lbuf, int lane)
{
    // low 32 bits of a generic pointer to __shared__ == LDS byte offset
    unsigned           loff = (unsigned)(unsigned long long)lbuf + lane * 16;
    unsigned long long ga   = (unsigned long long)gsrc + lane * 16;
    #pragma unroll
    for (int it = 0; it < 16; ++it) {
        asm volatile("global_load_async_to_lds_b128 %0, %1, off offset:%2"
                     :: "v"(loff), "v"(ga), "i"(it * 512)
                     : "memory");
    }
}

// ---------------------------------------------------------------------------
// Kernel 3: fused WMMA GEMM + logits + log-softmax.
// Grid: 256 blocks (one 16-row block each), 256 threads = 8 waves.
// Each wave owns column tiles jt = wave, wave+8, ...  Double-buffered async
// B staging with COMPILE-TIME buffer indices (loop unrolled by 2) so the
// fragment reads stay ds_load_b128 (addrspace(3) statically provable).
// ---------------------------------------------------------------------------
__global__ __launch_bounds__(256)
void logits_kernel(const _Float16* __restrict__ expL,
                   const _Float16* __restrict__ expR,
                   const float* __restrict__ mL,
                   const float* __restrict__ mR,
                   const float* __restrict__ mMax,
                   const float* __restrict__ temp,
                   float* __restrict__ out)
{
    const int tid  = threadIdx.x;
    const int wave = tid >> 5;        // 0..7
    const int lane = tid & 31;
    const int hk   = lane >> 4;       // lane half: 0 or 1
    const int l16  = lane & 15;
    const int i0   = blockIdx.x * 16; // first output row of this block

    __shared__ _Float16 ldsA[16 * FDIM];           //   8 KB
    __shared__ _Float16 ldsB[8][2][16 * FDIM];     // 128 KB, per-wave dbl buffer
    __shared__ float    mLsh[16];
    __shared__ float    cshift[16];
    __shared__ float    rowPart[8][16];
    __shared__ float    dRow[16];

    const float invtemp = 1.0f / temp[0];

    // ---- stage A block (16 rows x 256 f16 = 8 KB) into LDS, coalesced ----
    {
        const uint4* src = (const uint4*)(expL + (size_t)i0 * FDIM);
        uint4*       dst = (uint4*)ldsA;
        dst[tid]       = src[tid];
        dst[tid + 256] = src[tid + 256];
    }
    if (tid < 16) {
        const float ml = mL[i0 + tid];
        mLsh[tid]   = ml;
        cshift[tid] = (ml + mMax[0]) * invtemp;
    }
    __syncthreads();

    // ---- build A fragments (8 K-steps) from LDS, kept in registers ----
    unsigned aF[8][8];
    #pragma unroll
    for (int s = 0; s < 8; ++s) {
        #pragma unroll
        for (int v = 0; v < 8; ++v) {
            const int k = s * 32 + ((v < 4) ? (hk * 8 + 2 * v)
                                            : (16 + hk * 8 + 2 * (v - 4)));
            aF[s][v] = *(const unsigned*)&ldsA[l16 * FDIM + k];
        }
    }

    float eAcc[8] = {0, 0, 0, 0, 0, 0, 0, 0};

    // one 16-col tile: 8x WMMA over K=256 + logits transform + rowsum accum.
    // bufIdx is a compile-time constant at every call site.
    auto do_tile = [&](int j0, int bufIdx) {
        const _Float16* myB = ldsB[wave][bufIdx];

        v8f c = {0.f, 0.f, 0.f, 0.f, 0.f, 0.f, 0.f, 0.f};
        #pragma unroll
        for (int s = 0; s < 8; ++s) {
            union { v16h v; unsigned u[8]; } b;
            #pragma unroll
            for (int v = 0; v < 8; ++v) {
                const int k = s * 32 + ((v < 4) ? (hk * 8 + 2 * v)
                                                : (16 + hk * 8 + 2 * (v - 4)));
                b.u[v] = *(const unsigned*)&myB[l16 * FDIM + k];
            }
            union { v16h v; unsigned u[8]; } a;
            #pragma unroll
            for (int v = 0; v < 8; ++v) a.u[v] = aF[s][v];

            c = __builtin_amdgcn_wmma_f32_16x16x32_f16(
                    false, a.v, false, b.v, (short)0, c, false, false);
        }

        // u = (ln S + mL + mR)/temp; rowsum += exp2((u - cshift)*log2e).
        // S >= ~0.25 always (positive addends, max term O(1)) -> no denorm
        // guards needed: raw v_log_f32 / v_exp_f32.
        const float mRj = mR[j0 + l16];
        #pragma unroll
        for (int r = 0; r < 8; ++r) {
            const int m = r + 8 * hk;                 // C layout row
            const float l2 = __builtin_amdgcn_logf(c[r]);      // log2(S)
            const float u  = fmaf(l2, LN2, mLsh[m] + mRj) * invtemp;
            out[(size_t)(i0 + m) * NUM_CLASSES + j0 + l16] = u;
            eAcc[r] += __builtin_amdgcn_exp2f((u - cshift[m]) * LOG2E);
        }
    };

    // =====================  PASS 1 (double-buffered, unrolled x2) ==========
    async_tile(expR + (size_t)(wave * 16) * FDIM, ldsB[wave][0], lane);

    #pragma unroll 1
    for (int jt = wave; jt < NUM_CLASSES / 16; jt += 16) {
        // prefetch tile jt+8 into buffer 1 (always exists: jt+8 <= 2047)
        async_tile(expR + (size_t)(jt + 8) * 16 * FDIM, ldsB[wave][1], lane);
        asm volatile("s_wait_asynccnt 16" ::: "memory");   // buffer 0 ready
        do_tile(jt * 16, 0);

        if (jt + 16 < NUM_CLASSES / 16) {
            async_tile(expR + (size_t)(jt + 16) * 16 * FDIM, ldsB[wave][0], lane);
            asm volatile("s_wait_asynccnt 16" ::: "memory"); // buffer 1 ready
        } else {
            asm volatile("s_wait_asynccnt 0" ::: "memory");
        }
        do_tile((jt + 8) * 16, 1);
    }

    // ---- reduce eAcc over the 16 column-lanes of each half-wave ----
    #pragma unroll
    for (int r = 0; r < 8; ++r) {
        float v = eAcc[r];
        #pragma unroll
        for (int off = 1; off < 16; off <<= 1)
            v += __shfl_xor(v, off, 16);
        eAcc[r] = v;
    }
    if (l16 == 0) {
        #pragma unroll
        for (int r = 0; r < 8; ++r)
            rowPart[wave][r + 8 * hk] = eAcc[r];
    }
    __threadfence();          // make pass-1 global stores visible block-wide
    __syncthreads();

    if (tid < 16) {
        float s = 0.0f;
        #pragma unroll
        for (int w = 0; w < 8; ++w) s += rowPart[w][tid];
        dRow[tid] = cshift[tid] + __logf(s);   // logsumexp of row i0+tid
    }
    __syncthreads();

    // =====================  PASS 2  =====================
    {
        float4* o4 = (float4*)(out + (size_t)i0 * NUM_CLASSES);
        const int total = 16 * NUM_CLASSES / 4;      // 131072 float4
        for (int p = tid; p < total; p += 256) {
            const int row = p >> 13;                 // 8192 float4 per row
            const float d = dRow[row];
            float4 v = o4[p];
            v.x -= d; v.y -= d; v.z -= d; v.w -= d;
            o4[p] = v;
        }
    }
}

// ---------------------------------------------------------------------------
// Launcher
// ---------------------------------------------------------------------------
extern "C" void kernel_launch(void* const* d_in, const int* in_sizes, int n_in,
                              void* d_out, int out_size, void* d_ws, size_t ws_size,
                              hipStream_t stream) {
    const float* start_emb  = (const float*)d_in[0];  // [4096, 128]
    const float* output_emb = (const float*)d_in[1];  // [32768, 128]
    const float* proj       = (const float*)d_in[2];  // [128, 256]
    const float* temp       = (const float*)d_in[3];  // [1]
    float*       out        = (float*)d_out;          // [4096, 32768]

    char* ws = (char*)d_ws;
    _Float16* expL = (_Float16*)(ws);                                   //  2 MB
    _Float16* expR = (_Float16*)(ws + (size_t)NUM_STARTS * FDIM * 2);   // 16 MB
    float*    mL   = (float*)(ws + (size_t)(NUM_STARTS + NUM_CLASSES) * FDIM * 2);
    float*    mR   = mL + NUM_STARTS;
    float*    mMax = mR + NUM_CLASSES;

    feat_kernel<<<NUM_STARTS,  256, 0, stream>>>(start_emb,  proj, expL, mL);
    feat_kernel<<<NUM_CLASSES, 256, 0, stream>>>(output_emb, proj, expR, mR);
    mmax_kernel<<<1, 256, 0, stream>>>(mR, mMax);
    logits_kernel<<<NUM_STARTS / 16, 256, 0, stream>>>(expL, expR, mL, mR,
                                                       mMax, temp, out);
}
